// ColBertPairwiseDistillLoss_9740985828095
// MI455X (gfx1250) — matrix operand order repaired and checked by reading, not compiled, not run on hardware
//
#include <hip/hip_runtime.h>
#include <hip/hip_fp16.h>

typedef __attribute__((ext_vector_type(16))) _Float16 v16h;
typedef __attribute__((ext_vector_type(8)))  float    v8f;
typedef __attribute__((ext_vector_type(4)))  float    v4f;

#define NB   64      // batch (queries == docs count)
#define NQ   32      // query tokens
#define SD   1024    // doc tokens
#define DK   128     // embedding dim
#define NCHUNK 8     // S split into 8 chunks of 128 tokens
#define CTOK 128     // tokens per chunk
#define CTILE 8      // 16-token tiles per chunk

// Workspace layout (floats):
//   scores  : [2][64][64]                       @ 0          (32 KB)
//   partial : [2][64 c][8 chunk][64 b][32 row]  @ 8192       (8 MB)
//   aFrag   : f16 [2][64 b][2 mh][4 kc][2 hi/lo][2 h][16 m][16 e] @ +2097152 fl (2 MB)
#define WS_PARTIAL_OFF 8192
#define WS_AFRAG_OFF   (8192 + 2097152)

__device__ __forceinline__ v8f wmma_f16(v16h a, v16h b, v8f c) {
    return __builtin_amdgcn_wmma_f32_16x16x32_f16(false, a, false, b,
                                                  (short)0, c, false, false);
}

// ---------------------------------------------------------------------------
// Pass 1: convert queries to WMMA-A-layout hi/lo fp16 fragments.
// A 16x32 f16 layout: lane l: row m=l&15; element e -> K = k0 + 8h + (e<8 ? e : e+8)
// i.e. runs [k0+8h, +8) and [k0+16+8h, +8).   One thread per (mh,kc,h,m).
// ---------------------------------------------------------------------------
__global__ __launch_bounds__(256) void qconv_kernel(
    const float* __restrict__ q_all, const float* __restrict__ tq_all,
    float* __restrict__ ws)
{
    const int b     = blockIdx.x;
    const int which = blockIdx.y;
    const int tid   = threadIdx.x;
    const int m  = tid & 15;
    const int h  = (tid >> 4) & 1;
    const int kc = (tid >> 5) & 3;
    const int mh = tid >> 7;

    const float* Q = (which ? tq_all : q_all) + (size_t)b * NQ * DK;
    const float* qrow = Q + (size_t)(mh * 16 + m) * DK;
    const int k0 = kc * 32;

    float f[16];
    *(v4f*)&f[0]  = *(const v4f*)(qrow + k0 + 8 * h);
    *(v4f*)&f[4]  = *(const v4f*)(qrow + k0 + 8 * h + 4);
    *(v4f*)&f[8]  = *(const v4f*)(qrow + k0 + 16 + 8 * h);
    *(v4f*)&f[12] = *(const v4f*)(qrow + k0 + 16 + 8 * h + 4);

    v16h hi, lo;
    #pragma unroll
    for (int e = 0; e < 16; ++e) {
        float x = f[e];
        _Float16 hh = (_Float16)x;
        hi[e] = hh;
        lo[e] = (_Float16)(x - (float)hh);
    }

    _Float16* aFrag = (_Float16*)(ws + WS_AFRAG_OFF);
    // per (which,b,mh,kc): 1024 f16 = [part 2][h 2][m 16][e 16]
    size_t base = ((((size_t)which * NB + b) * 2 + mh) * 4 + kc) * 1024;
    size_t off  = (size_t)(h * 16 + m) * 16;
    *(v16h*)(aFrag + base + off)       = hi;   // part 0 = hi
    *(v16h*)(aFrag + base + 512 + off) = lo;   // part 1 = lo
}

// ---------------------------------------------------------------------------
// Pass 2: block = (c, S-chunk, tensor). Convert doc chunk ONCE into LDS
// B-fragments, then each of 8 waves sweeps 8 query batches; both row-halves
// are processed together so each LDS B-fragment load feeds 6 WMMAs.
// B 32x16 f16 layout: lane l: col n=l&15; element e -> K = k0 + 8h + (e<8 ? e : e+8)
// LDS layout: [tile 8][kc 4][h 2][n 16][e 16]  (hi block, then lo block)
// ---------------------------------------------------------------------------
__global__ __launch_bounds__(256) void maxsim_chunk_kernel(
    const float* __restrict__ d_all, const float* __restrict__ td_all,
    float* __restrict__ ws)
{
    __shared__ _Float16 ldsB[2 * 16384];   // 64 KB: hi @0, lo @16384

    const int c     = blockIdx.x >> 3;
    const int chunk = blockIdx.x & 7;
    const int which = blockIdx.y;

    const float* Dc = (which ? td_all : d_all)
                      + ((size_t)c * SD + (size_t)chunk * CTOK) * DK;

    const int tid  = threadIdx.x;
    const int wave = tid >> 5;
    const int lane = tid & 31;
    const int h    = lane >> 4;
    const int l15  = lane & 15;

    // ---- stage + convert doc chunk into LDS fragments (once per block) ----
    #pragma unroll
    for (int it = 0; it < 4; ++it) {
        const int pos = tid + it * 256;          // 0..1023 = (tile,kc,h,n)
        const int n  = pos & 15;
        const int hh = (pos >> 4) & 1;
        const int kc = (pos >> 5) & 3;
        const int t  = pos >> 7;
        const float* src = Dc + (size_t)(t * 16 + n) * DK + kc * 32 + 8 * hh;
        float f[16];
        *(v4f*)&f[0]  = *(const v4f*)(src);
        *(v4f*)&f[4]  = *(const v4f*)(src + 4);
        *(v4f*)&f[8]  = *(const v4f*)(src + 16);
        *(v4f*)&f[12] = *(const v4f*)(src + 20);
        v16h hi, lo;
        #pragma unroll
        for (int e = 0; e < 16; ++e) {
            float x = f[e];
            _Float16 v = (_Float16)x;
            hi[e] = v;
            lo[e] = (_Float16)(x - (float)v);
        }
        const int off = pos * 16;                // == ((t*4+kc)*32 + hh*16 + n)*16
        *(v16h*)&ldsB[off]         = hi;
        *(v16h*)&ldsB[16384 + off] = lo;
    }
    __syncthreads();

    const _Float16* aFrag = (const _Float16*)(ws + WS_AFRAG_OFF);
    float* partial = ws + WS_PARTIAL_OFF
                   + ((size_t)(which * NB + c) * NCHUNK + chunk) * (NB * NQ);

    // ---- each wave: 8 query batches, both row-halves per B-fragment load ----
    #pragma unroll 1
    for (int bi = 0; bi < 8; ++bi) {
        const int b = wave * 8 + bi;
        const size_t aoff = (size_t)(h * 16 + l15) * 16;
        const _Float16* abase = aFrag + ((size_t)which * NB + b) * 2 * 4096;

        v16h ah[2][4], al[2][4];               // A frags, both row-halves
        #pragma unroll
        for (int mh = 0; mh < 2; ++mh) {
            #pragma unroll
            for (int kc = 0; kc < 4; ++kc) {
                ah[mh][kc] = *(const v16h*)(abase + mh * 4096 + kc * 1024 + aoff);
                al[mh][kc] = *(const v16h*)(abase + mh * 4096 + kc * 1024 + 512 + aoff);
            }
        }

        float runmax[2][8];
        #pragma unroll
        for (int r = 0; r < 8; ++r) { runmax[0][r] = -3.0e38f; runmax[1][r] = -3.0e38f; }

        #pragma unroll 1
        for (int t = 0; t < CTILE; ++t) {
            v8f acc0 = {}, acc1 = {};
            #pragma unroll
            for (int kc = 0; kc < 4; ++kc) {
                const int boff = ((t * 4 + kc) * 32 + h * 16 + l15) * 16;
                v16h bh = *(const v16h*)&ldsB[boff];
                v16h bl = *(const v16h*)&ldsB[16384 + boff];
                acc0 = wmma_f16(ah[0][kc], bh, acc0);   // hi*hi
                acc0 = wmma_f16(ah[0][kc], bl, acc0);   // hi*lo
                acc0 = wmma_f16(al[0][kc], bh, acc0);   // lo*hi
                acc1 = wmma_f16(ah[1][kc], bh, acc1);
                acc1 = wmma_f16(ah[1][kc], bl, acc1);
                acc1 = wmma_f16(al[1][kc], bh, acc1);
            }
            #pragma unroll
            for (int r = 0; r < 8; ++r) {
                runmax[0][r] = fmaxf(runmax[0][r], acc0[r]);
                runmax[1][r] = fmaxf(runmax[1][r], acc1[r]);
            }
        }

        // max over 16 doc columns (lanes within each 16-lane half), then store
        #pragma unroll
        for (int mh = 0; mh < 2; ++mh) {
            #pragma unroll
            for (int r = 0; r < 8; ++r) {
                float v = runmax[mh][r];
                v = fmaxf(v, __shfl_xor(v, 1, 32));
                v = fmaxf(v, __shfl_xor(v, 2, 32));
                v = fmaxf(v, __shfl_xor(v, 4, 32));
                v = fmaxf(v, __shfl_xor(v, 8, 32));
                runmax[mh][r] = v;
            }
            // D layout: VGPR r + lane-half h -> query row r+8h
            if (l15 == 0) {
                #pragma unroll
                for (int r = 0; r < 8; ++r)
                    partial[(size_t)b * NQ + mh * 16 + r + 8 * h] = runmax[mh][r];
            }
        }
    }
}

// ---------------------------------------------------------------------------
// Pass 3: scores[which][b][c] = sum_rows max_chunks partial
// ---------------------------------------------------------------------------
__global__ __launch_bounds__(256) void combine_kernel(float* __restrict__ ws)
{
    const int t = blockIdx.x * 256 + threadIdx.x;   // 0..8191
    const int which = t >> 12;
    const int r = t & 4095;
    const int b = r >> 6;
    const int c = r & 63;

    const float* P = ws + WS_PARTIAL_OFF
                   + (size_t)(which * NB + c) * NCHUNK * (NB * NQ)
                   + (size_t)b * NQ;
    float s = 0.0f;
    for (int row = 0; row < NQ; ++row) {
        float m = P[row];
        #pragma unroll
        for (int ch = 1; ch < NCHUNK; ++ch)
            m = fmaxf(m, P[(size_t)ch * (NB * NQ) + row]);
        s += m;
    }
    ws[(size_t)which * NB * NB + b * NB + c] = s;
}

// ---------------------------------------------------------------------------
// Pass 4: hardest-negative softplus contrastive + fp16 MSE distillation.
// ---------------------------------------------------------------------------
__global__ __launch_bounds__(64) void loss_kernel(const float* __restrict__ ws,
                                                  float* __restrict__ out)
{
    const float* S = ws;
    const float* T = ws + NB * NB;
    const int b = threadIdx.x;

    float pos = 0.0f, neg = -3.0e38f, mse = 0.0f;
    for (int c = 0; c < NB; ++c) {
        float s = S[b * NB + c];
        float t = T[b * NB + c];
        __half d  = __hsub(__float2half(s), __float2half(t));   // fp16 arithmetic
        __half d2 = __hmul(d, d);
        mse += __half2float(d2);
        if (c == b) pos = s; else neg = fmaxf(neg, s);
    }
    float x  = neg - pos;
    float sp = fmaxf(x, 0.0f) + log1pf(expf(-fabsf(x)));        // stable softplus

    __shared__ float sbuf[NB], mbuf[NB];
    sbuf[b] = sp;
    mbuf[b] = mse;
    __syncthreads();
    if (b == 0) {
        float cs = 0.0f, ms = 0.0f;
        for (int i = 0; i < NB; ++i) { cs += sbuf[i]; ms += mbuf[i]; }
        out[0] = cs / (float)NB + 0.3f * (ms / (float)(NB * NB));
    }
}

extern "C" void kernel_launch(void* const* d_in, const int* in_sizes, int n_in,
                              void* d_out, int out_size, void* d_ws, size_t ws_size,
                              hipStream_t stream) {
    const float* q  = (const float*)d_in[0];
    const float* d  = (const float*)d_in[1];
    const float* tq = (const float*)d_in[2];
    const float* td = (const float*)d_in[3];
    float* ws = (float*)d_ws;

    qconv_kernel<<<dim3(NB, 2), 256, 0, stream>>>(q, tq, ws);
    maxsim_chunk_kernel<<<dim3(NB * NCHUNK, 2), 256, 0, stream>>>(d, td, ws);
    combine_kernel<<<32, 256, 0, stream>>>(ws);
    loss_kernel<<<1, NB, 0, stream>>>(ws, (float*)d_out);
}
